// CameraEncoder_60670708023400
// MI455X (gfx1250) — matrix operand order repaired and checked by reading, not compiled, not run on hardware
//
#include <hip/hip_runtime.h>
#include <hip/hip_bf16.h>
#include <math.h>

typedef __attribute__((ext_vector_type(16))) _Float16 v16h;
typedef __attribute__((ext_vector_type(8)))  float    v8f;
typedef unsigned int __attribute__((ext_vector_type(4))) uint4v;
typedef unsigned int __attribute__((ext_vector_type(8))) uint8v;

#define BB     2
#define NCAM   6
#define CC     64
#define FH_    16
#define FW_    44
#define NPIX   704        // FH_*FW_
#define DBINS  64
#define K1     576        // 9*CC   (K-order: k = (ky*3+kx)*64 + ic)
#define PADH   18
#define PADW   46
#define PADSZ  (PADH*PADW)   // 828
#define NIMG   12
#define BEVH   200
#define BEVW   200
#define PXT    64            // pixels per splat tile
#define NPT    (NPIX/PXT)    // 11

// ---- dynamic LDS partition for kernel 1 (bytes) ----
#define OFF_W1 0
#define SZ_W1  (CC*K1*2)          // 73728   [oc][k], k=(ky*3+kx)*64+ic
#define OFF_F  (OFF_W1 + SZ_W1)
#define SZ_F   (PADSZ*CC*2)       // 105984  [pady][padx][c]  (channel innermost)
#define OFF_W2 (OFF_F + SZ_F)
#define SZ_W2  (CC*CC*2)          // 8192    [d][c]
#define OFF_H  (OFF_W2 + SZ_W2)
#define SZ_H   (8*16*CC*2)        // 16384   per-wave [p][c]  (channel innermost)
#define OFF_SC (OFF_H + SZ_H)
#define SZ_SC  (CC*2*4)           // 512
#define LDS_TOTAL (OFF_SC + SZ_SC) // 204800 bytes (< 320KB WGP LDS)

// A-fragment (16-bit, 16x32) per-lane K mapping (contiguous, matches B):
// K = g*16 + e
__device__ __forceinline__ int a_kmap(int g, int e) {
    return (g << 4) + e;
}

// ---------------------------------------------------------------------------
// TDM: 2D tile load Global -> LDS (D# per CDNA5 ISA ch.8; 4-byte elements).
// Issued once by the calling wave; tracked with TENSORcnt.
// ---------------------------------------------------------------------------
__device__ __forceinline__ void tdm_load_2d_f32(void* lds, const void* gptr,
                                                unsigned tensor_d0, unsigned tensor_d1,
                                                unsigned tile_d0,   unsigned tile_d1,
                                                unsigned stride0)
{
    unsigned      lo = (unsigned)(size_t)lds;        // LDS aperture: addr[31:0] = LDS offset
    unsigned long ga = (unsigned long)(size_t)gptr;  // tile start address

    uint4v g0;
    g0.x = 1u;                                       // count=1, user-mode descriptor
    g0.y = lo;                                       // lds_addr [63:32]
    g0.z = (unsigned)ga;                             // global_addr[31:0]  -> bits [95:64]
    g0.w = (unsigned)(ga >> 32) | (2u << 30);        // global_addr[56:32], type=2 [127:126]

    uint8v g1;
    g1[0] = (2u << 16);                              // workgroup_mask=0, data_size=2 (4B)
    g1[1] = (tensor_d0 & 0xFFFFu) << 16;             // tensor_dim0 [79:48]
    g1[2] = (tensor_d0 >> 16) | ((tensor_d1 & 0xFFFFu) << 16); // tensor_dim1 [111:80]
    g1[3] = (tensor_d1 >> 16) | ((tile_d0 & 0xFFFFu) << 16);   // tile_dim0 [127:112]
    g1[4] = (tile_d1 & 0xFFFFu);                     // tile_dim1 [143:128], tile_dim2=0
    g1[5] = stride0;                                 // tensor_dim0_stride[31:0] [207:160]
    g1[6] = 0u;                                      // stride0[47:32]=0, dim1_stride lo=0
    g1[7] = 0u;                                      // dim1_stride hi (unused, 2D tile)

    asm volatile("tensor_load_to_lds %0, %1" :: "s"(g0), "s"(g1) : "memory");
}

// ---------------------------------------------------------------------------
// Kernel 1: per-image fused conv3x3 + BN + ReLU + conv1x1 + depth softmax
// ---------------------------------------------------------------------------
__global__ void __launch_bounds__(256)
lss_conv_depth_kernel(const float* __restrict__ feat,
                      const float* __restrict__ w1,  const float* __restrict__ b1,
                      const float* __restrict__ gamma, const float* __restrict__ beta,
                      const float* __restrict__ rmean, const float* __restrict__ rvar,
                      const float* __restrict__ w2,  const float* __restrict__ b2,
                      float* __restrict__ probs)
{
    extern __shared__ char smem[];
    _Float16* sW1 = (_Float16*)(smem + OFF_W1);   // [oc][(ky*3+kx)*64+ic]
    _Float16* sF  = (_Float16*)(smem + OFF_F);    // [(y*PADW+x)*CC + c]
    _Float16* sW2 = (_Float16*)(smem + OFF_W2);   // [d][c]
    _Float16* sH  = (_Float16*)(smem + OFF_H);    // per-wave [p][c]
    float* sScale = (float*)(smem + OFF_SC);
    float* sShift = sScale + CC;

    const int img = blockIdx.x;
    const int tid = threadIdx.x;

    // stage w1 as f16, re-ordering K: src k_old = ic*9 + r  ->  dst k = r*64 + ic
    for (int j = tid; j < CC*K1; j += 256) {
        int oc  = j / K1;
        int rem = j - oc*K1;
        int ic  = rem / 9;
        int r   = rem - ic*9;
        sW1[oc*K1 + r*CC + ic] = (_Float16)w1[j];
    }
    for (int i = tid; i < CC*CC; i += 256) sW2[i] = (_Float16)w2[i];
    if (tid < CC) {
        float sc = gamma[tid] * rsqrtf(rvar[tid] + 1e-5f);
        sScale[tid] = sc;
        sShift[tid] = (b1[tid] - rmean[tid]) * sc + beta[tid]; // fold conv bias + BN
    }
    // zero-padded f16 feature image, channel innermost
    const float* fimg = feat + (size_t)img * (CC*NPIX);
    for (int i = tid; i < PADSZ*CC; i += 256) {
        int c = i & (CC-1);
        int p = i >> 6;
        int y = p / PADW, x = p - y*PADW;
        _Float16 v = (_Float16)0.f;
        if (y >= 1 && y <= FH_ && x >= 1 && x <= FW_)
            v = (_Float16)fimg[c*NPIX + (y-1)*FW_ + (x-1)];
        sF[i] = v;
    }
    __syncthreads();

    const int wave = tid >> 5;
    const int lane = tid & 31;
    const int g    = lane >> 4;   // lane-group (0: lanes 0-15, 1: lanes 16-31)
    const int ln   = lane & 15;
    _Float16* sHw  = sH + wave * (16*CC);

    for (int pt = wave; pt < 44; pt += 8) {          // 44 pixel tiles of 16
        const int pix = pt*16 + ln;
        const int py  = pix / FW_, px = pix - py*FW_;

        // ---- conv1 GEMM: M=64 (4 tiles), K=576 (18 steps of 32) ----
        v8f acc[4] = {v8f{}, v8f{}, v8f{}, v8f{}};
        for (int ks = 0; ks < 18; ++ks) {
            // B fragment (32x16): each 16-run has fixed (ky,kx), contiguous ic
            const int k0  = ks*32 + g*16;
            const int r   = k0 >> 6;            // = ky*3+kx (fixed for this run)
            const int icb = k0 & 63;
            const int ky  = r / 3, kx = r - ky*3;
            const _Float16* bsrc = sF + ((py+ky)*PADW + (px+kx))*CC + icb;
            v16h bf;
            #pragma unroll
            for (int e = 0; e < 16; ++e) bf[e] = bsrc[e];   // 2x ds_load_b128

            #pragma unroll
            for (int mt = 0; mt < 4; ++mt) {
                v16h af;
                int row = mt*16 + ln;
                #pragma unroll
                for (int e = 0; e < 16; ++e)
                    af[e] = sW1[row*K1 + ks*32 + a_kmap(g, e)];
                acc[mt] = __builtin_amdgcn_wmma_f32_16x16x32_f16(
                    false, af, false, bf, (short)0, acc[mt], false, false);
            }
        }

        // ---- BN + ReLU epilogue; h -> per-wave LDS [p][c] (contiguous c runs) ----
        // C-layout: acc[mt] reg r holds (m = mt*16 + g*8 + r, n = ln)
        #pragma unroll
        for (int mt = 0; mt < 4; ++mt)
            #pragma unroll
            for (int r = 0; r < 8; ++r) {
                int c = mt*16 + g*8 + r;
                float h = acc[mt][r] * sScale[c] + sShift[c];
                sHw[ln*CC + c] = (_Float16)fmaxf(h, 0.f);
            }
        // same-wave DS ops are in-order; no barrier needed for sHw reuse

        // ---- conv2 GEMM: logits(64 x 16) = W2(64x64) @ h(64x16) ----
        v8f lacc[4] = {v8f{}, v8f{}, v8f{}, v8f{}};
        #pragma unroll
        for (int ks = 0; ks < 2; ++ks) {
            const _Float16* bsrc2 = sHw + ln*CC + ks*32 + g*16;
            v16h bf;
            #pragma unroll
            for (int e = 0; e < 16; ++e) bf[e] = bsrc2[e];  // 2x ds_load_b128
            #pragma unroll
            for (int mt = 0; mt < 4; ++mt) {
                v16h af;
                int row = mt*16 + ln;
                #pragma unroll
                for (int e = 0; e < 16; ++e)
                    af[e] = sW2[row*CC + ks*32 + a_kmap(g, e)];
                lacc[mt] = __builtin_amdgcn_wmma_f32_16x16x32_f16(
                    false, af, false, bf, (short)0, lacc[mt], false, false);
            }
        }

        // ---- softmax over 64 depth bins (d split between lane p and p+16) ----
        float lg[32];
        #pragma unroll
        for (int mt = 0; mt < 4; ++mt)
            #pragma unroll
            for (int r = 0; r < 8; ++r) {
                int d = mt*16 + g*8 + r;
                lg[mt*8 + r] = lacc[mt][r] + b2[d];
            }
        float mx = lg[0];
        #pragma unroll
        for (int i = 1; i < 32; ++i) mx = fmaxf(mx, lg[i]);
        mx = fmaxf(mx, __shfl_xor(mx, 16, 32));
        float s = 0.f;
        #pragma unroll
        for (int i = 0; i < 32; ++i) { lg[i] = __expf(lg[i] - mx); s += lg[i]; }
        s += __shfl_xor(s, 16, 32);
        float inv = 1.f / s;

        float* pimg = probs + (size_t)img * (DBINS*NPIX);
        #pragma unroll
        for (int mt = 0; mt < 4; ++mt)
            #pragma unroll
            for (int r = 0; r < 8; ++r) {
                int d = mt*16 + g*8 + r;
                pimg[d*NPIX + pix] = lg[mt*8 + r] * inv;
            }
    }
}

// ---------------------------------------------------------------------------
// Kernel 2: frustum projection + BEV scatter.
// One block per (image, 64-pixel tile). TDM stages feat/prob tiles into LDS
// overlapped with the geometry pass; scatter does L2-resident f32 atomics.
// ---------------------------------------------------------------------------
__global__ void __launch_bounds__(256)
lss_splat_kernel(const float* __restrict__ feat,
                 const float* __restrict__ intrin,
                 const float* __restrict__ extrin,
                 const float* __restrict__ probs,
                 float* __restrict__ out)
{
    __shared__ float sFeat[CC*PXT];      // [c][px]
    __shared__ float sProb[DBINS*PXT];   // [d][px]
    __shared__ int   sBin [DBINS*PXT];   // bev cell (sans channel) or -1

    const int blk = blockIdx.x;
    const int img = blk / NPT;
    const int pt  = blk - img*NPT;
    const int b   = img / NCAM;
    const int tid = threadIdx.x;

    // ---- wave 0: kick off both tile DMAs (TENSORcnt), overlap with geometry ----
    if (tid < 32) {
        tdm_load_2d_f32(sFeat, feat  + (size_t)img*CC*NPIX    + pt*PXT,
                        NPIX, CC,    PXT, CC,    NPIX);
        tdm_load_2d_f32(sProb, probs + (size_t)img*DBINS*NPIX + pt*PXT,
                        NPIX, DBINS, PXT, DBINS, NPIX);
    }

    // ---- geometry pass: bin index for all (d, px) of this tile ----
    const float* K = intrin + (size_t)img * 9;
    float a00=K[0],a01=K[1],a02=K[2],a10=K[3],a11=K[4],a12=K[5],a20=K[6],a21=K[7],a22=K[8];
    float det = a00*(a11*a22-a12*a21) - a01*(a10*a22-a12*a20) + a02*(a10*a21-a11*a20);
    float id  = 1.f / det;
    float i00=(a11*a22-a12*a21)*id, i01=(a02*a21-a01*a22)*id, i02=(a01*a12-a02*a11)*id;
    float i10=(a12*a20-a10*a22)*id, i11=(a00*a22-a02*a20)*id, i12=(a02*a10-a00*a12)*id;
    float i20=(a10*a21-a11*a20)*id, i21=(a01*a20-a00*a21)*id, i22=(a00*a11-a01*a10)*id;
    const float* E = extrin + (size_t)img * 16;

    for (int idx = tid; idx < DBINS*PXT; idx += 256) {
        int di = idx >> 6;                 // depth bin
        int px = idx & (PXT-1);
        int pix = pt*PXT + px;
        int fy = pix / FW_, fx = pix - fy*FW_;

        float d  = 1.0f + (float)di * (59.0f / 63.0f);   // linspace(1,60,64)
        float xi = (float)fx * (703.0f / 43.0f);         // linspace(0,703,44)
        float yi = (float)fy * 17.0f;                    // linspace(0,255,16)
        float ppx = xi * d, ppy = yi * d, ppz = d;

        float cx = i00*ppx + i01*ppy + i02*ppz;
        float cy = i10*ppx + i11*ppy + i12*ppz;
        float cz = i20*ppx + i21*ppy + i22*ppz;
        float ex = E[0]*cx + E[1]*cy + E[2]*cz + E[3];
        float ey = E[4]*cx + E[5]*cy + E[6]*cz + E[7];

        int ix = (int)floorf((ex + 50.f) * 2.f);         // res = 0.5
        int iy = (int)floorf((ey + 50.f) * 2.f);
        int ok = (ix >= 0) & (ix < BEVW) & (iy >= 0) & (iy < BEVH);
        sBin[idx] = ok ? (b*(CC*BEVH*BEVW) + iy*BEVW + ix) : -1;
    }

    if (tid < 32) __builtin_amdgcn_s_wait_tensorcnt(0);  // TDM tiles landed
    __syncthreads();

    // ---- scatter: 64-thread channel group per (d, px) pair ----
    const int c  = tid & 63;
    const int cOff = c * (BEVH*BEVW);
    for (int idx = tid >> 6; idx < DBINS*PXT; idx += 4) {
        int bin = sBin[idx];
        if (bin < 0) continue;
        int di = idx >> 6;
        int px = idx & (PXT-1);
        float v = sFeat[c*PXT + px] * sProb[di*PXT + px];
        atomicAdd(&out[bin + cOff], v);
    }
}

__global__ void zero_kernel(float* __restrict__ out, int n) {
    int i = blockIdx.x * blockDim.x + threadIdx.x;
    if (i < n) out[i] = 0.f;
}

extern "C" void kernel_launch(void* const* d_in, const int* in_sizes, int n_in,
                              void* d_out, int out_size, void* d_ws, size_t ws_size,
                              hipStream_t stream) {
    const float* features = (const float*)d_in[0];   // (B,N,C,FH,FW)
    const float* intrin   = (const float*)d_in[1];   // (B,N,3,3)
    const float* extrin   = (const float*)d_in[2];   // (B,N,4,4)
    const float* w1    = (const float*)d_in[3];
    const float* b1    = (const float*)d_in[4];
    const float* gamma = (const float*)d_in[5];
    const float* beta  = (const float*)d_in[6];
    const float* rmean = (const float*)d_in[7];
    const float* rvar  = (const float*)d_in[8];
    const float* w2    = (const float*)d_in[9];
    const float* b2    = (const float*)d_in[10];

    float* out   = (float*)d_out;                    // (B,C,200,200)
    float* probs = (float*)d_ws;                     // [12][64][704] f32 (~2.1 MB)

    zero_kernel<<<(out_size + 255) / 256, 256, 0, stream>>>(out, out_size);

    lss_conv_depth_kernel<<<NIMG, 256, LDS_TOTAL, stream>>>(
        features, w1, b1, gamma, beta, rmean, rvar, w2, b2, probs);

    lss_splat_kernel<<<NIMG * NPT, 256, 0, stream>>>(
        features, intrin, extrin, probs, out);
}
